// MoE_ConvBlock_32452772888759
// MI455X (gfx1250) — matrix operand order repaired and checked by reading, not compiled, not run on hardware
//
#include <hip/hip_runtime.h>
#include <stdint.h>

// ---------------------------------------------------------------------------
// MoE ConvBlock for MI455X (gfx1250, wave32).
// Precision strategy: bf16 WMMA (v_wmma_f32_16x16x32_bf16) with fp32
// accumulate for all 1x1-conv GEMMs; depthwise 3x3 in fp32 VALU (mem-bound).
// Only the 2 routed experts are computed (reference computes all 8).
// ---------------------------------------------------------------------------

#define CCH   128
#define HW    4096
#define WIMG  64
#define HIMG  64
#define NB    16
#define NEXP  8
#define TDIM  256
#define CDIM  256
#define TSTR  136   // padded LDS row stride (bf16 elems): 68 dwords -> bank step 4

typedef __bf16 v16bf __attribute__((ext_vector_type(16)));
typedef float  v8f   __attribute__((ext_vector_type(8)));

union BF16x16 { uint4 q[2]; v16bf v; };

__device__ __forceinline__ unsigned short f2bf(float f) {
  unsigned u = __float_as_uint(f);
  u += 0x7FFFu + ((u >> 16) & 1u);     // round-to-nearest-even
  return (unsigned short)(u >> 16);
}
__device__ __forceinline__ float silu_f(float x) { return x / (1.0f + __expf(-x)); }

__device__ __forceinline__ v8f wmma_bf16(v16bf a, v16bf b, v8f c) {
  return __builtin_amdgcn_wmma_f32_16x16x32_bf16(false, a, false, b, (short)0, c,
                                                 false, false);
}

// A fragment: W[co][ci] tile, M=co, K=ci.  Lanes 0-15: K {0..7,16..23};
// lanes 16-31: K {8..15,24..31} of row m0+(lane&15).  16B-aligned b128 loads.
__device__ __forceinline__ v16bf load_frag_A(const unsigned short* Wl, int m0, int k0,
                                             int lane) {
  const unsigned short* p = Wl + (m0 + (lane & 15)) * TSTR + k0 + ((lane >> 4) << 3);
  BF16x16 u;
  u.q[0] = *(const uint4*)(p);
  u.q[1] = *(const uint4*)(p + 16);
  return u.v;
}

// B fragment: T[hw][ci] tile (K=ci contiguous per column).  Lanes 0-15: K {0..7,16..23},
// lanes 16-31: K {8..15,24..31} of column n0+(lane&15); two 16B loads per lane.
__device__ __forceinline__ v16bf load_frag_B(const unsigned short* Tl, int n0, int k0,
                                             int lane) {
  const unsigned short* p = Tl + (n0 + (lane & 15)) * TSTR + k0 + ((lane >> 4) << 3);
  BF16x16 u;
  u.q[0] = *(const uint4*)(p);
  u.q[1] = *(const uint4*)(p + 16);
  return u.v;
}

struct alignas(16) LdsBlk {
  unsigned short T[WIMG * TSTR];   // dw output tile, [hw][ci] bf16  (17408 B)
  unsigned short W[CCH * TSTR];    // pw weights,     [co][ci] bf16  (34816 B)
  float dwW[CCH * 9];              // depthwise 3x3 weights          ( 4608 B)
  float bdw[CCH];
  float bpw[CCH];
};                                  // total 57856 B < 64 KB

// Depthwise 3x3 ('SAME') over one image row h0, all 128 channels, bf16 -> LDS.
__device__ __forceinline__ void dw_phase(const float* __restrict__ src, int h0, int tid,
                                         unsigned short* T, const float* dwW,
                                         const float* bdw) {
  for (int it = 0; it < 32; ++it) {
    int ci = (it << 2) | (tid >> 6);
    int w  = tid & 63;
    const float* xc = src + (size_t)ci * HW + (size_t)h0 * WIMG;
    const float* wd = dwW + ci * 9;
    float s = bdw[ci];
#pragma unroll
    for (int dh = -1; dh <= 1; ++dh) {
      int h = h0 + dh;
      if (h >= 0 && h < HIMG) {
        const float* xr = xc + dh * WIMG;
        float v0 = (w > 0)      ? xr[w - 1] : 0.0f;
        float v1 = xr[w];
        float v2 = (w < 63)     ? xr[w + 1] : 0.0f;
        const float* wr = wd + (dh + 1) * 3;
        s += v0 * wr[0] + v1 * wr[1] + v2 * wr[2];
      }
    }
    T[w * TSTR + ci] = f2bf(s);
  }
}

// ---------------------------------------------------------------------------
// Kernel 1: proj = pw(dw(x)), fused; writes x_proj fp32 NCHW to workspace.
// grid (64 rows, 16 images), 256 threads (8 waves).
// ---------------------------------------------------------------------------
__global__ __launch_bounds__(256) void k_proj(const float* __restrict__ x,
                                              const float* __restrict__ dw_w,
                                              const float* __restrict__ dw_b,
                                              const float* __restrict__ pw_w,
                                              const float* __restrict__ pw_b,
                                              float* __restrict__ xproj) {
  __shared__ LdsBlk S;
  const int tid = threadIdx.x;
  const int b   = blockIdx.y;
  const int h0  = blockIdx.x;

  for (int i = tid; i < CCH * CCH; i += 256)
    S.W[(i >> 7) * TSTR + (i & 127)] = f2bf(pw_w[i]);
  for (int i = tid; i < CCH * 9; i += 256) S.dwW[i] = dw_w[i];
  if (tid < CCH) { S.bdw[tid] = dw_b[tid]; S.bpw[tid] = pw_b[tid]; }
  __syncthreads();

  const float* xb = x + (size_t)b * CCH * HW;
  dw_phase(xb, h0, tid, S.T, S.dwW, S.bdw);
  __syncthreads();

  const int lane   = tid & 31;
  const int m0     = (tid >> 5) << 4;          // wave -> co tile
  const int rowadd = (lane >> 4) << 3;

  v16bf a0 = load_frag_A(S.W, m0, 0,  lane);
  v16bf a1 = load_frag_A(S.W, m0, 32, lane);
  v16bf a2 = load_frag_A(S.W, m0, 64, lane);
  v16bf a3 = load_frag_A(S.W, m0, 96, lane);

  v8f cinit;
#pragma unroll
  for (int r = 0; r < 8; ++r) cinit[r] = S.bpw[m0 + rowadd + r];

  float* outp = xproj + (size_t)b * CCH * HW + (size_t)h0 * WIMG;
#pragma unroll
  for (int n = 0; n < 4; ++n) {
    int n0 = n << 4;
    v8f acc = cinit;
    acc = wmma_bf16(a0, load_frag_B(S.T, n0, 0,  lane), acc);
    acc = wmma_bf16(a1, load_frag_B(S.T, n0, 32, lane), acc);
    acc = wmma_bf16(a2, load_frag_B(S.T, n0, 64, lane), acc);
    acc = wmma_bf16(a3, load_frag_B(S.T, n0, 96, lane), acc);
    int colw = n0 + (lane & 15);
#pragma unroll
    for (int r = 0; r < 8; ++r)
      outp[(size_t)(m0 + rowadd + r) * HW + colw] = acc[r];
  }
}

// ---------------------------------------------------------------------------
// Kernel 2: global average pool.  One block per (b, c).
// ---------------------------------------------------------------------------
__global__ __launch_bounds__(256) void k_pool(const float* __restrict__ xproj,
                                              float* __restrict__ pooled) {
  __shared__ float red[256];
  const int id = blockIdx.x;                   // b*128 + c
  const float* p = xproj + (size_t)id * HW;
  float s = 0.0f;
  for (int i = threadIdx.x; i < HW; i += 256) s += p[i];
  red[threadIdx.x] = s;
  __syncthreads();
  for (int st = 128; st > 0; st >>= 1) {
    if (threadIdx.x < st) red[threadIdx.x] += red[threadIdx.x + st];
    __syncthreads();
  }
  if (threadIdx.x == 0) pooled[id] = red[0] * (1.0f / HW);
}

// ---------------------------------------------------------------------------
// Kernel 3: router MLP + top-2 softmax.  One block per image, 128 threads.
// ---------------------------------------------------------------------------
__global__ __launch_bounds__(128) void k_router(const float* __restrict__ pooled,
                                                const float* __restrict__ t_emb,
                                                const float* __restrict__ c_emb,
                                                const float* __restrict__ w1,
                                                const float* __restrict__ b1,
                                                const float* __restrict__ w2,
                                                const float* __restrict__ b2,
                                                float* __restrict__ logits_out,
                                                float* __restrict__ routing) {
  __shared__ float feat[CCH + TDIM + CDIM];
  __shared__ float hbuf[CCH];
  __shared__ float lg[NEXP];
  const int b = blockIdx.x, tid = threadIdx.x;
  feat[tid]       = pooled[b * CCH + tid];
  feat[128 + tid] = t_emb[b * TDIM + tid];
  feat[256 + tid] = t_emb[b * TDIM + 128 + tid];
  feat[384 + tid] = c_emb[b * CDIM + tid];
  feat[512 + tid] = c_emb[b * CDIM + 128 + tid];
  __syncthreads();
  float s = b1[tid];
  for (int f = 0; f < CCH + TDIM + CDIM; ++f) s += feat[f] * w1[f * CCH + tid];
  hbuf[tid] = silu_f(s);
  __syncthreads();
  if (tid < NEXP) {
    float t = b2[tid];
    for (int j = 0; j < CCH; ++j) t += hbuf[j] * w2[j * NEXP + tid];
    lg[tid] = t;
    logits_out[b * NEXP + tid] = t;
  }
  __syncthreads();
  if (tid == 0) {
    int i0 = 0; float v0 = lg[0];
    for (int e = 1; e < NEXP; ++e) if (lg[e] > v0) { v0 = lg[e]; i0 = e; }
    int i1 = (i0 == 0) ? 1 : 0; float v1 = lg[i1];
    for (int e = 0; e < NEXP; ++e)
      if (e != i0 && lg[e] > v1) { v1 = lg[e]; i1 = e; }
    float e0 = 1.0f;                       // exp(v0 - v0)
    float e1 = __expf(v1 - v0);
    float inv = 1.0f / (e0 + e1);
    routing[b * 4 + 0] = (float)i0;
    routing[b * 4 + 1] = (float)i1;
    routing[b * 4 + 2] = e0 * inv;
    routing[b * 4 + 3] = e1 * inv;
  }
}

// ---------------------------------------------------------------------------
// Kernel 4: the 2 routed experts, fused dw -> WMMA pw -> silu -> weighted sum.
// grid (64 rows, 16 images), 256 threads.
// ---------------------------------------------------------------------------
__global__ __launch_bounds__(256) void k_experts(const float* __restrict__ xproj,
                                                 const float* __restrict__ routing,
                                                 const float* __restrict__ edw,
                                                 const float* __restrict__ edwb,
                                                 const float* __restrict__ epw,
                                                 const float* __restrict__ epwb,
                                                 float* __restrict__ out) {
  __shared__ LdsBlk S;
  const int tid    = threadIdx.x;
  const int b      = blockIdx.y;
  const int h0     = blockIdx.x;
  const int lane   = tid & 31;
  const int m0     = (tid >> 5) << 4;
  const int rowadd = (lane >> 4) << 3;

  v8f oacc[4];
#pragma unroll
  for (int n = 0; n < 4; ++n)
#pragma unroll
    for (int r = 0; r < 8; ++r) oacc[n][r] = 0.0f;

  const float* xb = xproj + (size_t)b * CCH * HW;

#pragma unroll
  for (int ke = 0; ke < 2; ++ke) {
    const int   e  = (int)routing[b * 4 + ke];
    const float we = routing[b * 4 + 2 + ke];
    __syncthreads();   // previous iteration's LDS consumers are done
    const float* pw  = epw + (size_t)e * (CCH * CCH);
    const float* dwp = edw + (size_t)e * (CCH * 9);
    for (int i = tid; i < CCH * CCH; i += 256)
      S.W[(i >> 7) * TSTR + (i & 127)] = f2bf(pw[i]);
    for (int i = tid; i < CCH * 9; i += 256) S.dwW[i] = dwp[i];
    if (tid < CCH) { S.bdw[tid] = edwb[e * CCH + tid]; S.bpw[tid] = epwb[e * CCH + tid]; }
    __syncthreads();

    dw_phase(xb, h0, tid, S.T, S.dwW, S.bdw);
    __syncthreads();

    v16bf a0 = load_frag_A(S.W, m0, 0,  lane);
    v16bf a1 = load_frag_A(S.W, m0, 32, lane);
    v16bf a2 = load_frag_A(S.W, m0, 64, lane);
    v16bf a3 = load_frag_A(S.W, m0, 96, lane);

    v8f cinit;
#pragma unroll
    for (int r = 0; r < 8; ++r) cinit[r] = S.bpw[m0 + rowadd + r];

#pragma unroll
    for (int n = 0; n < 4; ++n) {
      int n0 = n << 4;
      v8f acc = cinit;
      acc = wmma_bf16(a0, load_frag_B(S.T, n0, 0,  lane), acc);
      acc = wmma_bf16(a1, load_frag_B(S.T, n0, 32, lane), acc);
      acc = wmma_bf16(a2, load_frag_B(S.T, n0, 64, lane), acc);
      acc = wmma_bf16(a3, load_frag_B(S.T, n0, 96, lane), acc);
#pragma unroll
      for (int r = 0; r < 8; ++r) oacc[n][r] += silu_f(acc[r]) * we;
    }
  }

  float* op = out + (size_t)b * CCH * HW + (size_t)h0 * WIMG;
#pragma unroll
  for (int n = 0; n < 4; ++n) {
    int colw = (n << 4) + (lane & 15);
#pragma unroll
    for (int r = 0; r < 8; ++r)
      op[(size_t)(m0 + rowadd + r) * HW + colw] = oacc[n][r];
  }
}

// ---------------------------------------------------------------------------
extern "C" void kernel_launch(void* const* d_in, const int* in_sizes, int n_in,
                              void* d_out, int out_size, void* d_ws, size_t ws_size,
                              hipStream_t stream) {
  (void)in_sizes; (void)n_in; (void)out_size; (void)ws_size;
  const float* x      = (const float*)d_in[0];
  const float* t_emb  = (const float*)d_in[1];
  const float* c_emb  = (const float*)d_in[2];
  const float* pdw    = (const float*)d_in[3];
  const float* pdwb   = (const float*)d_in[4];
  const float* ppw    = (const float*)d_in[5];
  const float* ppwb   = (const float*)d_in[6];
  const float* rw1    = (const float*)d_in[7];
  const float* rb1    = (const float*)d_in[8];
  const float* rw2    = (const float*)d_in[9];
  const float* rb2    = (const float*)d_in[10];
  const float* edw    = (const float*)d_in[11];
  const float* edwb   = (const float*)d_in[12];
  const float* epw    = (const float*)d_in[13];
  const float* epwb   = (const float*)d_in[14];

  float* out = (float*)d_out;
  const size_t xproj_elems = (size_t)NB * CCH * HW;   // 8,388,608 floats (32 MB)

  float* wsf     = (float*)d_ws;
  float* xproj   = wsf;
  float* pooled  = wsf + xproj_elems;                 // 2048 floats
  float* routing = pooled + NB * CCH;                 // 64 floats

  dim3 grid(HIMG, NB);
  k_proj<<<grid, 256, 0, stream>>>(x, pdw, pdwb, ppw, ppwb, xproj);
  k_pool<<<NB * CCH, 256, 0, stream>>>(xproj, pooled);
  k_router<<<NB, 128, 0, stream>>>(pooled, t_emb, c_emb, rw1, rb1, rw2, rb2,
                                   out + xproj_elems, routing);
  k_experts<<<grid, 256, 0, stream>>>(xproj, routing, edw, edwb, epw, epwb, out);
}
